// TemporalTransformer_82987358093443
// MI455X (gfx1250) — matrix operand order repaired and checked by reading, not compiled, not run on hardware
//
#include <hip/hip_runtime.h>
#include <hip/hip_bf16.h>
#include <math.h>

// ---------------------------------------------------------------------------
// CDNA5 (gfx1250) wave32 transformer forward.
// Matrix math: v_wmma_f32_16x16x32_bf16 (bf16 in, f32 accumulate).
// Tile staging: global_load_async_to_lds_b128 (ASYNCcnt), double-buffered so
// the async copy of K-slab i+1 overlaps the WMMAs of K-slab i.
// ---------------------------------------------------------------------------

typedef __bf16 bf16_t;
typedef __bf16 v16bf __attribute__((ext_vector_type(16)));
typedef __bf16 v8bf  __attribute__((ext_vector_type(8)));
typedef float  v8f   __attribute__((ext_vector_type(8)));

#define WMMA_BF16(a, b, c) \
  __builtin_amdgcn_wmma_f32_16x16x32_bf16(false, (a), false, (b), (short)0, (c), false, false)

// Async copy 16B global -> LDS (tracked by ASYNCcnt, not DScnt/LOADcnt).
__device__ __forceinline__ void async_ld_b128(void* lds_dst, const void* gsrc) {
  const unsigned int l = (unsigned int)(uintptr_t)lds_dst;  // low 32b = LDS offset
  asm volatile("global_load_async_to_lds_b128 %0, %1, off"
               :
               : "v"(l), "v"(gsrc)
               : "memory");
}
__device__ __forceinline__ void wait_async0() {
  asm volatile("s_wait_asynccnt 0" ::: "memory");
}

// Assemble a v16bf fragment from two 16B loads (LDS or global).
__device__ __forceinline__ v16bf ld_frag2x8(const bf16_t* p0, const bf16_t* p1) {
  v8bf lo = *(const v8bf*)p0;
  v8bf hi = *(const v8bf*)p1;
  v16bf r;
#pragma unroll
  for (int i = 0; i < 8; ++i) { r[i] = lo[i]; r[i + 8] = hi[i]; }
  return r;
}

__device__ __forceinline__ float gelu_exact(float x) {
  return 0.5f * x * (1.0f + erff(x * 0.7071067811865475f));
}

// ---------------------------------------------------------------------------
// One-time conversion kernels (off the hot path).
// ---------------------------------------------------------------------------
__global__ __launch_bounds__(256) void cast_bf16_k(const float* __restrict__ src,
                                                   bf16_t* __restrict__ dst, int n) {
  const int i = blockIdx.x * 256 + threadIdx.x;
  if (i < n) dst[i] = (bf16_t)src[i];
}

// Wt[n*K + k] = (bf16) W[k*N + n]  -- tiled 32x32 transpose-cast.
__global__ __launch_bounds__(256) void transpose_cast_k(const float* __restrict__ W,
                                                        bf16_t* __restrict__ Wt,
                                                        int K, int N) {
  __shared__ float tile[32][33];
  const int n0 = blockIdx.x * 32, k0 = blockIdx.y * 32;
  const int tx = threadIdx.x, ty = threadIdx.y;  // (32, 8)
#pragma unroll
  for (int i = 0; i < 4; ++i)
    tile[ty * 4 + i][tx] = W[(size_t)(k0 + ty * 4 + i) * N + (n0 + tx)];
  __syncthreads();
#pragma unroll
  for (int i = 0; i < 4; ++i)
    Wt[(size_t)(n0 + ty * 4 + i) * K + (k0 + tx)] = (bf16_t)tile[tx][ty * 4 + i];
}

// ---------------------------------------------------------------------------
// GEMM: C[M,N] = epi(A[M,K] @ Wt[N,K]^T + bias [, +R])
// A, Wt bf16 (k-contiguous). Block 256 thr / 8 waves, tile 128x128, K-step 32.
// Wave (wm=w&3, wn=w>>2): 32(M) x 64(N) -> 2x4 WMMA accumulators.
// Double-buffered LDS: async copy of slab i+1 overlaps WMMAs of slab i.
// MODE: 0 = bias, 1 = bias+GELU(exact), 2 = bias+residual. OUTBF: bf16 output.
// ---------------------------------------------------------------------------
template <int MODE, bool OUTBF>
__global__ __launch_bounds__(256) void gemm_k(
    const bf16_t* __restrict__ A, const bf16_t* __restrict__ Wt,
    const float* __restrict__ bias, const float* __restrict__ R,
    float* __restrict__ Cf, bf16_t* __restrict__ Cb, int M, int N, int K) {
  __shared__ __align__(32) bf16_t As[2][128][32];  // [buf][m][k]
  __shared__ __align__(32) bf16_t Bs[2][128][32];  // [buf][n][k]

  const int m0 = blockIdx.y * 128, n0 = blockIdx.x * 128;
  const int tid = threadIdx.x, lane = tid & 31, wave = tid >> 5;
  const int half = lane >> 4, l16 = lane & 15;
  const int wm = wave & 3, wn = wave >> 2;

  v8f acc[2][4] = {};

  const int srow = tid >> 1;        // 0..127
  const int sseg = (tid & 1) * 16;  // 0 or 16 (bf16 elems)

  auto stage = [&](int buf, int k0) {
    const bf16_t* ag = A + (size_t)(m0 + srow) * K + (k0 + sseg);
    const bf16_t* bg = Wt + (size_t)(n0 + srow) * K + (k0 + sseg);
    async_ld_b128(&As[buf][srow][sseg], ag);
    async_ld_b128(&As[buf][srow][sseg + 8], ag + 8);
    async_ld_b128(&Bs[buf][srow][sseg], bg);
    async_ld_b128(&Bs[buf][srow][sseg + 8], bg + 8);
  };

  // prologue: fill buffer 0
  stage(0, 0);
  wait_async0();
  __syncthreads();

  const int nsteps = K >> 5;
  for (int it = 0; it < nsteps; ++it) {
    const int buf = it & 1;
    if (it + 1 < nsteps) stage(buf ^ 1, (it + 1) << 5);  // overlaps WMMAs below

    // A frag (16x32): lane M=l16, elem e -> k = e + 16*half (contiguous)
    v16bf af[2];
#pragma unroll
    for (int i = 0; i < 2; ++i) {
      const bf16_t* p = &As[buf][wm * 32 + i * 16 + l16][half * 16];
      af[i] = ld_frag2x8(p, p + 8);
    }
    // B frag (32x16): lane N=l16, elem e -> k = e + 16*half (contiguous)
#pragma unroll
    for (int j = 0; j < 4; ++j) {
      const bf16_t* p = &Bs[buf][wn * 64 + j * 16 + l16][half * 16];
      v16bf bb = ld_frag2x8(p, p + 8);
#pragma unroll
      for (int i = 0; i < 2; ++i) acc[i][j] = WMMA_BF16(af[i], bb, acc[i][j]);
    }

    wait_async0();    // next slab landed
    __syncthreads();  // all waves done reading/writing this round
  }

  // Epilogue. C layout: lane N=l16, VGPR r -> M = r + 8*half.
#pragma unroll
  for (int i = 0; i < 2; ++i) {
#pragma unroll
    for (int j = 0; j < 4; ++j) {
      const int gn = n0 + wn * 64 + j * 16 + l16;
      const float bv = bias[gn];
#pragma unroll
      for (int r = 0; r < 8; ++r) {
        const int gm = m0 + wm * 32 + i * 16 + half * 8 + r;
        float v = acc[i][j][r] + bv;
        if (MODE == 1) v = gelu_exact(v);
        if (MODE == 2) v += R[(size_t)gm * N + gn];
        if (OUTBF)
          Cb[(size_t)gm * N + gn] = (bf16_t)v;
        else
          Cf[(size_t)gm * N + gn] = v;
      }
    }
  }
}

// ---------------------------------------------------------------------------
// LayerNorm over D=512 (f32 in, bf16 out). One wave per row, 8 rows / block.
// ---------------------------------------------------------------------------
__global__ __launch_bounds__(256) void layernorm_k(
    const float* __restrict__ x, const float* __restrict__ g,
    const float* __restrict__ b, bf16_t* __restrict__ y) {
  const int wave = threadIdx.x >> 5, lane = threadIdx.x & 31;
  const int row = blockIdx.x * 8 + wave;
  const float* xr = x + (size_t)row * 512;
  float v[16];
  float s = 0.0f;
#pragma unroll
  for (int i = 0; i < 16; ++i) { v[i] = xr[lane + i * 32]; s += v[i]; }
#pragma unroll
  for (int off = 16; off >= 1; off >>= 1) s += __shfl_xor(s, off, 32);
  const float mu = s * (1.0f / 512.0f);
  float var = 0.0f;
#pragma unroll
  for (int i = 0; i < 16; ++i) { const float d = v[i] - mu; var += d * d; }
#pragma unroll
  for (int off = 16; off >= 1; off >>= 1) var += __shfl_xor(var, off, 32);
  const float inv = rsqrtf(var * (1.0f / 512.0f) + 1e-5f);
  bf16_t* yr = y + (size_t)row * 512;
#pragma unroll
  for (int i = 0; i < 16; ++i) {
    const int c = lane + i * 32;
    yr[c] = (bf16_t)((v[i] - mu) * inv * g[c] + b[c]);
  }
}

// ---------------------------------------------------------------------------
// Sinusoidal PE add (f32 residual stream). D=512, T=2048.
// ---------------------------------------------------------------------------
__global__ __launch_bounds__(256) void add_pe_k(float* __restrict__ hbuf) {
  const int idx = blockIdx.x * 256 + threadIdx.x;
  const int d = idx & 511;
  const int t = (idx >> 9) & 2047;
  const float e = expf(-(float)(d & ~1) * (9.210340371976184f / 512.0f));  // ln(1e4)/D
  const float a = (float)t * e;
  hbuf[idx] += (d & 1) ? cosf(a) : sinf(a);
}

// ---------------------------------------------------------------------------
// Causal flash attention, bf16 WMMA. qkv bf16 [B*T,1536], out bf16 [B*T,512].
// Grid (T/64, B*H); block 128 = 4 waves; wave owns a 16-row Q tile.
// Reference ALiBi bias reduces exactly to {0 for k<=q, -1e30 for k>q}.
// ---------------------------------------------------------------------------
__global__ __launch_bounds__(128) void attn_k(const bf16_t* __restrict__ qkv,
                                              bf16_t* __restrict__ out) {
  const int Tn = 2048, Dn = 512, TRI = 1536;
  const int bh = blockIdx.y;
  const int b = bh >> 3, h = bh & 7;
  const int qblk = blockIdx.x;
  const int tid = threadIdx.x;
  const int lane = tid & 31, wave = tid >> 5;
  const int half = lane >> 4, l16 = lane & 15;
  const int qbase = qblk * 64 + wave * 16;

  const bf16_t* Qg = qkv + (size_t)b * Tn * TRI + h * 64;
  const bf16_t* Kg = Qg + 512;
  const bf16_t* Vg = Qg + 1024;

  __shared__ __align__(32) bf16_t Ks[32][64];     // K chunk [key][hd]
  __shared__ __align__(32) bf16_t Vts[64][32];    // V chunk transposed [hd][key]
  __shared__ __align__(32) bf16_t Ps[4][16][32];  // per-wave P staging [q][key]

  // Q fragments (A layout), hd chunks 0..31 / 32..63 -- direct bf16 loads
  v16bf qa[2];
  {
    const bf16_t* qp = Qg + (size_t)(qbase + l16) * TRI;
#pragma unroll
    for (int c = 0; c < 2; ++c) {
      const bf16_t* p = qp + c * 32 + half * 16;
      qa[c] = ld_frag2x8(p, p + 8);
    }
  }

  v8f acc[4] = {};
  float mrow[8], lrow[8];
#pragma unroll
  for (int r = 0; r < 8; ++r) { mrow[r] = -3.0e38f; lrow[r] = 0.0f; }

  const int nch = qblk * 2 + 2;  // key chunks up to the block's causal horizon
  for (int ch = 0; ch < nch; ++ch) {
    const int kb = ch * 32;
    {  // stage K (async bf16 copy) and V^T (needs reshaping -> LDS stores)
      const int kr = tid >> 2;
      const int seg = (tid & 3) * 16;
      const bf16_t* kp = Kg + (size_t)(kb + kr) * TRI + seg;
      async_ld_b128(&Ks[kr][seg], kp);
      async_ld_b128(&Ks[kr][seg + 8], kp + 8);
      const bf16_t* vp = Vg + (size_t)(kb + kr) * TRI + seg;
      v8bf v0 = *(const v8bf*)vp, v1 = *(const v8bf*)(vp + 8);
#pragma unroll
      for (int i = 0; i < 8; ++i) {
        Vts[seg + i][kr] = v0[i];
        Vts[seg + 8 + i][kr] = v1[i];
      }
      wait_async0();
    }
    __syncthreads();

    // S = Q K^T : two 16x16 key tiles, K-dim = HD = 64 (2 WMMAs each)
    v8f s[2] = {};
#pragma unroll
    for (int j = 0; j < 2; ++j)
#pragma unroll
      for (int c = 0; c < 2; ++c) {
        const bf16_t* p = &Ks[j * 16 + l16][c * 32 + half * 16];
        v16bf bb = ld_frag2x8(p, p + 8);
        s[j] = WMMA_BF16(qa[c], bb, s[j]);
      }

    // scale + causal mask (faithful: logits*scale + (-1e30 if k>q))
    const float scale = 0.125f;  // 1/sqrt(64)
    float sv[2][8];
#pragma unroll
    for (int j = 0; j < 2; ++j) {
      const int key = kb + j * 16 + l16;
#pragma unroll
      for (int r = 0; r < 8; ++r) {
        const int q = qbase + half * 8 + r;
        float xx = s[j][r] * scale;
        if (key > q) xx += -1.0e30f;
        sv[j][r] = xx;
      }
    }

    // online softmax; row stats via 16-lane butterflies (masks 8..1 stay in half)
    float p[2][8];
#pragma unroll
    for (int r = 0; r < 8; ++r) {
      float m = fmaxf(sv[0][r], sv[1][r]);
#pragma unroll
      for (int off = 8; off >= 1; off >>= 1) m = fmaxf(m, __shfl_xor(m, off, 32));
      const float mnew = fmaxf(mrow[r], m);
      const float p0 = __expf(sv[0][r] - mnew);
      const float p1 = __expf(sv[1][r] - mnew);
      p[0][r] = p0;
      p[1][r] = p1;
      float rs = p0 + p1;
#pragma unroll
      for (int off = 8; off >= 1; off >>= 1) rs += __shfl_xor(rs, off, 32);
      const float corr = __expf(mrow[r] - mnew);
      lrow[r] = lrow[r] * corr + rs;
      mrow[r] = mnew;
#pragma unroll
      for (int jh = 0; jh < 4; ++jh) acc[jh][r] *= corr;
    }

    // C-layout -> A-layout for P via per-wave LDS buffer
#pragma unroll
    for (int j = 0; j < 2; ++j)
#pragma unroll
      for (int r = 0; r < 8; ++r)
        Ps[wave][half * 8 + r][j * 16 + l16] = (bf16_t)p[j][r];
    __syncthreads();

    v16bf pa;
    {
      const bf16_t* pp = &Ps[wave][l16][half * 16];
      pa = ld_frag2x8(pp, pp + 8);
    }

    // O += P V : 4 hd tiles of 16
#pragma unroll
    for (int jh = 0; jh < 4; ++jh) {
      const bf16_t* vp2 = &Vts[jh * 16 + l16][half * 16];
      v16bf vb = ld_frag2x8(vp2, vp2 + 8);
      acc[jh] = WMMA_BF16(pa, vb, acc[jh]);
    }
    __syncthreads();  // before next chunk restages Ks/Vts
  }

  // normalize, write attnY (bf16) at [b,t, h*64+hd]
#pragma unroll
  for (int r = 0; r < 8; ++r) {
    const float inv = 1.0f / lrow[r];
    const int q = qbase + half * 8 + r;
    bf16_t* op = out + (size_t)(b * Tn + q) * Dn + h * 64;
#pragma unroll
    for (int jh = 0; jh < 4; ++jh) op[jh * 16 + l16] = (bf16_t)(acc[jh][r] * inv);
  }
}

// ---------------------------------------------------------------------------
// Host-side driver
// ---------------------------------------------------------------------------
static inline void run_gemm(const bf16_t* A, const bf16_t* Wt, const float* bias,
                            const float* R, float* Cf, bf16_t* Cb, int M, int N,
                            int K, int mode, hipStream_t s) {
  dim3 g(N / 128, M / 128), b(256);
  if (mode == 0 && Cb)
    gemm_k<0, true><<<g, b, 0, s>>>(A, Wt, bias, R, Cf, Cb, M, N, K);
  else if (mode == 0)
    gemm_k<0, false><<<g, b, 0, s>>>(A, Wt, bias, R, Cf, Cb, M, N, K);
  else if (mode == 1)
    gemm_k<1, true><<<g, b, 0, s>>>(A, Wt, bias, R, Cf, Cb, M, N, K);
  else
    gemm_k<2, false><<<g, b, 0, s>>>(A, Wt, bias, R, Cf, Cb, M, N, K);
}

static inline void run_tp(const float* W, bf16_t* Wt, int K, int N, hipStream_t s) {
  transpose_cast_k<<<dim3(N / 32, K / 32), dim3(32, 8), 0, s>>>(W, Wt, K, N);
}

extern "C" void kernel_launch(void* const* d_in, const int* in_sizes, int n_in,
                              void* d_out, int out_size, void* d_ws, size_t ws_size,
                              hipStream_t stream) {
  (void)in_sizes; (void)n_in; (void)out_size; (void)ws_size;

  const float* x    = (const float*)d_in[0];
  const float* Win  = (const float*)d_in[1];
  const float* bin_ = (const float*)d_in[2];
  const float* Wqkv = (const float*)d_in[3];
  const float* bqkv = (const float*)d_in[4];
  const float* Wo   = (const float*)d_in[5];
  const float* bo   = (const float*)d_in[6];
  const float* ln1g = (const float*)d_in[7];
  const float* ln1b = (const float*)d_in[8];
  const float* Wf1  = (const float*)d_in[9];
  const float* bf1  = (const float*)d_in[10];
  const float* Wf2  = (const float*)d_in[11];
  const float* bf2  = (const float*)d_in[12];
  const float* ln2g = (const float*)d_in[13];
  const float* ln2b = (const float*)d_in[14];
  const float* lnfg = (const float*)d_in[15];
  const float* lnfb = (const float*)d_in[16];
  const float* Wout = (const float*)d_in[17];
  const float* bout = (const float*)d_in[18];
  float* out = (float*)d_out;

  const int M = 4096;  // B*T

  // ---- workspace layout ----
  char* ws = (char*)d_ws;
  size_t off = 0;
  float*  h     = (float*)(ws + off);  off += (size_t)M * 512 * 4;      // 8 MiB
  bf16_t* y     = (bf16_t*)(ws + off); off += (size_t)M * 512 * 2;     // 4 MiB
  bf16_t* big   = (bf16_t*)(ws + off); off += (size_t)M * 2048 * 2;    // 16 MiB
  bf16_t* xbf   = (bf16_t*)(ws + off); off += (size_t)M * 128 * 2;     // 1 MiB
  bf16_t* WinT  = (bf16_t*)(ws + off); off += (size_t)512 * 128 * 2;
  bf16_t* WoutT = (bf16_t*)(ws + off); off += (size_t)128 * 512 * 2;
  bf16_t* WqkvT = (bf16_t*)(ws + off); off += (size_t)4 * 1536 * 512 * 2;
  bf16_t* WoT   = (bf16_t*)(ws + off); off += (size_t)4 * 512 * 512 * 2;
  bf16_t* Wf1T  = (bf16_t*)(ws + off); off += (size_t)4 * 2048 * 512 * 2;
  bf16_t* Wf2T  = (bf16_t*)(ws + off); off += (size_t)4 * 512 * 2048 * 2;

  // ---- one-time casts / weight transposes (bf16, k-contiguous) ----
  cast_bf16_k<<<(M * 128) / 256, 256, 0, stream>>>(x, xbf, M * 128);
  run_tp(Win, WinT, 128, 512, stream);
  run_tp(Wout, WoutT, 512, 128, stream);
  for (int l = 0; l < 4; ++l) {
    run_tp(Wqkv + (size_t)l * 512 * 1536, WqkvT + (size_t)l * 1536 * 512, 512, 1536, stream);
    run_tp(Wo + (size_t)l * 512 * 512, WoT + (size_t)l * 512 * 512, 512, 512, stream);
    run_tp(Wf1 + (size_t)l * 512 * 2048, Wf1T + (size_t)l * 2048 * 512, 512, 2048, stream);
    run_tp(Wf2 + (size_t)l * 2048 * 512, Wf2T + (size_t)l * 512 * 2048, 2048, 512, stream);
  }

  // ---- forward ----
  run_gemm(xbf, WinT, bin_, nullptr, h, nullptr, M, 512, 128, 0, stream);
  add_pe_k<<<(M * 512) / 256, 256, 0, stream>>>(h);

  for (int l = 0; l < 4; ++l) {
    layernorm_k<<<512, 256, 0, stream>>>(h, ln1g + l * 512, ln1b + l * 512, y);
    run_gemm(y, WqkvT + (size_t)l * 1536 * 512, bqkv + l * 1536, nullptr,
             nullptr, big, M, 1536, 512, 0, stream);
    attn_k<<<dim3(32, 16), 128, 0, stream>>>(big, y);
    run_gemm(y, WoT + (size_t)l * 512 * 512, bo + l * 512, h, h, nullptr,
             M, 512, 512, 2, stream);
    layernorm_k<<<512, 256, 0, stream>>>(h, ln2g + l * 512, ln2b + l * 512, y);
    run_gemm(y, Wf1T + (size_t)l * 2048 * 512, bf1 + l * 2048, nullptr,
             nullptr, big, M, 2048, 512, 1, stream);
    run_gemm(big, Wf2T + (size_t)l * 512 * 2048, bf2 + l * 512, h, h, nullptr,
             M, 512, 2048, 2, stream);
  }

  layernorm_k<<<512, 256, 0, stream>>>(h, lnfg, lnfb, y);
  run_gemm(y, WoutT, bout, nullptr, out, nullptr, M, 128, 512, 0, stream);
}
